// SingleHeadAttention_13743895347373
// MI455X (gfx1250) — matrix operand order repaired and checked by reading, not compiled, not run on hardware
//
#include <hip/hip_runtime.h>

// ---------------------------------------------------------------------------
// Single-head causal attention for MI455X (gfx1250), bf16 WMMA pipeline:
//   1) cast x -> bf16; cast+transpose Wq/Wk/Wv -> bf16 [out][in]
//   2) WMMA GEMM: Q = (x Wq)/32 (bf16 rowmajor), K = x Wk (bf16 rowmajor),
//      Vt = (x Wv)^T per batch (bf16, [b][e][s])
//   3) WMMA GEMM: scores = Q K^T (f32), skipping causally-dead 128x128 tiles
//   4) causal softmax (wave-per-row, wave32 shuffles) -> P bf16
//   5) WMMA GEMM: out = P V  (K-loop truncated at m0+128 by causality)
// GEMM uses GLOBAL_LOAD_ASYNC_TO_LDS_B128 (ASYNCcnt) double-buffered staging.
// ---------------------------------------------------------------------------

typedef unsigned short u16;
typedef unsigned int   u32;
typedef __attribute__((ext_vector_type(4)))  unsigned int v4u;
typedef __attribute__((ext_vector_type(8)))  unsigned int v8u;
typedef __attribute__((ext_vector_type(16))) __bf16      v16bf;
typedef __attribute__((ext_vector_type(8)))  float       v8f;

__device__ __forceinline__ u16 f32_to_bf16(float f) {
  u32 u = __float_as_uint(f);
  u32 r = u + 0x7FFFu + ((u >> 16) & 1u);   // round to nearest even
  return (u16)(r >> 16);
}

// Build a v16bf WMMA operand from two 16B LDS chunks (ds_load_b128 x2).
__device__ __forceinline__ v16bf load_frag(const u16* p0, const u16* p1) {
  v4u lo = *(const v4u*)p0;
  v4u hi = *(const v4u*)p1;
  v8u t  = __builtin_shufflevector(lo, hi, 0, 1, 2, 3, 4, 5, 6, 7);
  return __builtin_bit_cast(v16bf, t);
}

// Async DMA: 16B global -> LDS, tracked by ASYNCcnt (CDNA5).
__device__ __forceinline__ void async_ld16(u16* lds_dst, const u16* gsrc) {
  u32 l = (u32)(uintptr_t)lds_dst;      // flat LDS-aperture low bits = DS addr
  asm volatile("global_load_async_to_lds_b128 %0, %1, off"
               :: "v"(l), "v"(gsrc) : "memory");
}
__device__ __forceinline__ void wait_async0() {
  asm volatile("s_wait_asynccnt 0" ::: "memory");
}

// ---------------------------------------------------------------------------
// Generic bf16 WMMA GEMM:  C[z] = A[z] (MxK, rowmajor) * Bt[z]^T
// Bt stored K-contiguous per output column: Bt[n][k], ldb.
// outMode: 0 = bf16 rowmajor (with scale)
//          1 = bf16 transposed per-MperB batch: Ct[m/MperB][n][m%MperB]
//          2 = f32 rowmajor
// causalMode: 1 = skip tile if n0 >= m0+BM (scores), 2 = Klim = m0+BM (P*V)
// ---------------------------------------------------------------------------
__global__ __launch_bounds__(256) void gemm_bf16_wmma(
    const u16* __restrict__ A, long lda, long abat,
    const u16* __restrict__ B, long ldb, long bbat,
    void* __restrict__ C, long ldc, long cbat,
    int M, int N, int K,
    int outMode, float scale, int causalMode, int MperB)
{
  constexpr int BM = 128, BN = 128, BK = 32;
  constexpr int ASTR = 48, BSTR = 48;           // padded LDS strides (bf16 units)
  __shared__ u16 As[2][BM * ASTR];              // 2 x 12 KB
  __shared__ u16 Bs[2][BN * BSTR];              // 2 x 12 KB

  const int m0 = blockIdx.y * BM;
  const int n0 = blockIdx.x * BN;
  const int z  = blockIdx.z;

  if (causalMode == 1 && n0 >= m0 + BM) return; // causally dead score tile
  int Klim = K;
  if (causalMode == 2) { int kl = m0 + BM; Klim = kl < K ? kl : K; }

  const u16* Ap = A + (long)z * abat;
  const u16* Bp = B + (long)z * bbat;

  const int tid  = threadIdx.x;
  const int lane = tid & 31;
  const int wid  = tid >> 5;
  const int wm   = (wid & 3) * 32;   // wave's row base within tile (4 groups)
  const int wn   = (wid >> 2) * 64;  // wave's col base within tile (2 groups)
  const int lrow = lane & 15;
  const int hlf  = lane >> 4;

  // stage one K-tile (BM x BK of A, BN x BK of B) via async DMA into buffer b
  auto stage = [&](int k0, int bufIdx) {
    u16* Ab = &As[bufIdx][0];
    u16* Bb = &Bs[bufIdx][0];
#pragma unroll
    for (int t = 0; t < 2; ++t) {               // 512 chunks each / 256 thr
      int c = tid + t * 256;
      int r = c >> 2, j = (c & 3) * 8;
      async_ld16(&Ab[r * ASTR + j], Ap + (long)(m0 + r) * lda + (k0 + j));
      async_ld16(&Bb[r * BSTR + j], Bp + (long)(n0 + r) * ldb + (k0 + j));
    }
  };

  v8f acc[2][4];
#pragma unroll
  for (int i = 0; i < 2; ++i)
#pragma unroll
    for (int j = 0; j < 4; ++j)
      acc[i][j] = (v8f){0.f, 0.f, 0.f, 0.f, 0.f, 0.f, 0.f, 0.f};

  const int nsteps = Klim / BK;
  stage(0, 0);
  wait_async0();
  __syncthreads();

  for (int s = 0; s < nsteps; ++s) {
    const int cur = s & 1;
    if (s + 1 < nsteps) stage((s + 1) * BK, cur ^ 1);  // overlap DMA w/ WMMA

    // ---- gather fragments per ISA 7.12.2 wave32 layouts ----
    v16bf af[2], bf[4];
#pragma unroll
    for (int i = 0; i < 2; ++i) {
      const u16* base = &As[cur][(wm + i * 16 + lrow) * ASTR + hlf * 8];
      af[i] = load_frag(base, base + 16);       // K = {h*8..+7, h*8+16..+23}
    }
#pragma unroll
    for (int j = 0; j < 4; ++j) {
      const u16* base = &Bs[cur][(wn + j * 16 + lrow) * BSTR + hlf * 8];
      bf[j] = load_frag(base, base + 16);       // K = {h*8..+7, h*8+16..+23}
    }
#pragma unroll
    for (int i = 0; i < 2; ++i)
#pragma unroll
      for (int j = 0; j < 4; ++j)
        acc[i][j] = __builtin_amdgcn_wmma_f32_16x16x32_bf16(
            false, af[i], false, bf[j], (short)0, acc[i][j], false, false);

    wait_async0();      // next buffer fully written (per-wave ASYNCcnt)
    __syncthreads();    // all waves done reading cur + all DMAs landed
  }

  // ---- store C fragments (lane: fixed N, 8 consecutive M per half) ----
  if (outMode == 0) {
    u16* Cp = (u16*)C + (long)z * cbat;
#pragma unroll
    for (int i = 0; i < 2; ++i)
#pragma unroll
      for (int j = 0; j < 4; ++j) {
        int n  = n0 + wn + j * 16 + lrow;
        int mb = m0 + wm + i * 16 + hlf * 8;
#pragma unroll
        for (int v = 0; v < 8; ++v)
          Cp[(long)(mb + v) * ldc + n] = f32_to_bf16(acc[i][j][v] * scale);
      }
  } else if (outMode == 1) {
    u16* Cp = (u16*)C;   // batched through the M index
#pragma unroll
    for (int i = 0; i < 2; ++i)
#pragma unroll
      for (int j = 0; j < 4; ++j) {
        int n  = n0 + wn + j * 16 + lrow;
        int mb = m0 + wm + i * 16 + hlf * 8;
        int bz = mb / MperB, mo = mb % MperB;
        u32 w[4];
#pragma unroll
        for (int v = 0; v < 8; v += 2)
          w[v >> 1] = (u32)f32_to_bf16(acc[i][j][v] * scale) |
                      ((u32)f32_to_bf16(acc[i][j][v + 1] * scale) << 16);
        v4u pk = {w[0], w[1], w[2], w[3]};
        *(v4u*)&Cp[(long)bz * cbat + (long)n * ldc + mo] = pk;  // 16B store
      }
  } else {
    float* Cp = (float*)C + (long)z * cbat;
#pragma unroll
    for (int i = 0; i < 2; ++i)
#pragma unroll
      for (int j = 0; j < 4; ++j) {
        int n  = n0 + wn + j * 16 + lrow;
        int mb = m0 + wm + i * 16 + hlf * 8;
#pragma unroll
        for (int v = 0; v < 8; ++v)
          Cp[(long)(mb + v) * ldc + n] = acc[i][j][v] * scale;
      }
  }
}

// ---------------------------------------------------------------------------
// Causal softmax: one wave32 per row. Reads f32 scores row [0..i], writes
// bf16 P, zero-padded out to the 128-row GEMM tile boundary.
// ---------------------------------------------------------------------------
__global__ __launch_bounds__(256) void softmax_causal(
    const float* __restrict__ Sc, u16* __restrict__ P, int S)
{
  const int wid = threadIdx.x >> 5, lane = threadIdx.x & 31;
  const long row = (long)blockIdx.x * 8 + wid;   // global row in [0, B*S)
  const int b = (int)(row / S), i = (int)(row % S);
  const float* s = Sc + ((long)b * S + i) * (long)S;
  u16*        p = P  + ((long)b * S + i) * (long)S;

  float mx = -3.402823466e38f;
  for (int k = lane; k <= i; k += 32) mx = fmaxf(mx, s[k]);
#pragma unroll
  for (int off = 16; off > 0; off >>= 1) mx = fmaxf(mx, __shfl_xor(mx, off, 32));

  float sum = 0.f;
  for (int k = lane; k <= i; k += 32) sum += __expf(s[k] - mx);
#pragma unroll
  for (int off = 16; off > 0; off >>= 1) sum += __shfl_xor(sum, off, 32);
  const float inv = 1.0f / sum;

  int lim = (i & ~127) + 128; if (lim > S) lim = S;  // cover P*V tile reads
  for (int k = lane; k < lim; k += 32) {
    float v = (k <= i) ? __expf(s[k] - mx) * inv : 0.f;
    p[k] = f32_to_bf16(v);
  }
}

// ---------------------------------------------------------------------------
__global__ void cvt_f32_bf16(const float* __restrict__ in, u16* __restrict__ out, int n) {
  int idx = blockIdx.x * blockDim.x + threadIdx.x;
  if (idx < n) out[idx] = f32_to_bf16(in[idx]);
}

// W [K][N] f32 -> Wt [N][K] bf16 (K-contiguous for WMMA B operands)
__global__ void cvt_transpose_f32_bf16(const float* __restrict__ W, u16* __restrict__ Wt,
                                       int K, int N) {
  int idx = blockIdx.x * blockDim.x + threadIdx.x;
  if (idx < K * N) {
    int k = idx / N, n = idx % N;
    Wt[(long)n * K + k] = f32_to_bf16(W[idx]);
  }
}

// ---------------------------------------------------------------------------
extern "C" void kernel_launch(void* const* d_in, const int* in_sizes, int n_in,
                              void* d_out, int out_size, void* d_ws, size_t ws_size,
                              hipStream_t stream) {
  (void)in_sizes; (void)n_in; (void)out_size; (void)ws_size;
  const int B = 4, S = 2048, D = 1024;
  const float* x  = (const float*)d_in[0];
  const float* Wq = (const float*)d_in[1];
  const float* Wk = (const float*)d_in[2];
  const float* Wv = (const float*)d_in[3];
  float* out = (float*)d_out;

  // workspace carve-up (~166 MB total)
  char* ws = (char*)d_ws;
  size_t o = 0;
  u16* xb  = (u16*)(ws + o); o += (size_t)B * S * D * 2;      // 16 MB
  u16* Wt[3];
  for (int w = 0; w < 3; ++w) { Wt[w] = (u16*)(ws + o); o += (size_t)D * D * 2; }
  u16* Qb  = (u16*)(ws + o); o += (size_t)B * S * D * 2;      // 16 MB
  u16* Kb  = (u16*)(ws + o); o += (size_t)B * S * D * 2;      // 16 MB
  u16* Vt  = (u16*)(ws + o); o += (size_t)B * S * D * 2;      // 16 MB ([b][e][s])
  float* Sc = (float*)(ws + o); o += (size_t)B * S * S * 4;   // 64 MB
  u16* P   = (u16*)(ws + o); o += (size_t)B * S * S * 2;      // 32 MB

  // 1) casts
  const int nx = B * S * D;
  cvt_f32_bf16<<<(nx + 255) / 256, 256, 0, stream>>>(x, xb, nx);
  cvt_transpose_f32_bf16<<<(D * D + 255) / 256, 256, 0, stream>>>(Wq, Wt[0], D, D);
  cvt_transpose_f32_bf16<<<(D * D + 255) / 256, 256, 0, stream>>>(Wk, Wt[1], D, D);
  cvt_transpose_f32_bf16<<<(D * D + 255) / 256, 256, 0, stream>>>(Wv, Wt[2], D, D);

  // 2) projections: M = B*S = 8192, N = K = D
  dim3 g1(D / 128, (B * S) / 128, 1);
  gemm_bf16_wmma<<<g1, 256, 0, stream>>>(xb, D, 0, Wt[0], D, 0, Qb, D, 0,
      B * S, D, D, /*out*/0, /*scale=1/sqrt(D)*/ 1.0f / 32.0f, /*causal*/0, S);
  gemm_bf16_wmma<<<g1, 256, 0, stream>>>(xb, D, 0, Wt[1], D, 0, Kb, D, 0,
      B * S, D, D, 0, 1.0f, 0, S);
  gemm_bf16_wmma<<<g1, 256, 0, stream>>>(xb, D, 0, Wt[2], D, 0, Vt, S, (long)D * S,
      B * S, D, D, /*transposed bf16*/1, 1.0f, 0, S);

  // 3) scores = Q K^T per batch (causal tile-skip), f32
  dim3 g2(S / 128, S / 128, B);
  gemm_bf16_wmma<<<g2, 256, 0, stream>>>(Qb, D, (long)S * D, Kb, D, (long)S * D,
      Sc, S, (long)S * S, S, S, D, /*f32*/2, 1.0f, /*causal skip*/1, S);

  // 4) causal softmax -> P (bf16)
  softmax_causal<<<(B * S) / 8, 256, 0, stream>>>(Sc, P, S);

  // 5) out = P V per batch (K-loop truncated by causality), f32
  dim3 g3(D / 128, S / 128, B);
  gemm_bf16_wmma<<<g3, 256, 0, stream>>>(P, S, (long)S * S, Vt, S, (long)D * S,
      out, D, (long)S * D, S, D, S, 2, 1.0f, /*K limit*/2, S);
}
